// Fingerprint_33191507264112
// MI455X (gfx1250) — compile-verified, matrix operand order checked by reading, not run on hardware
//
#include <hip/hip_runtime.h>
#include <math.h>

// ---------------------------------------------------------------------------
// Types for CDNA5 WMMA (wave32): v_wmma_f32_16x16x32_bf16
// ---------------------------------------------------------------------------
typedef __attribute__((ext_vector_type(16))) __bf16 v16bf;
typedef __attribute__((ext_vector_type(8)))  float  v8f;
typedef __attribute__((ext_vector_type(4)))  unsigned int v4u;

// TDM descriptor vector types (6-arg clang-23 form)
typedef __attribute__((ext_vector_type(4))) unsigned int tdm_v4u;
typedef __attribute__((ext_vector_type(8))) int          tdm_v8i;
typedef __attribute__((ext_vector_type(4))) int          tdm_v4i;

union Frag {
    v16bf v;
    unsigned int u[8];
    v4u q[2];
};

__device__ __forceinline__ unsigned short f2bf(float f) {
    union { float f; unsigned int u; } a;
    a.f = f;
    unsigned int r = a.u + 0x7FFFu + ((a.u >> 16) & 1u); // round-to-nearest-even
    return (unsigned short)(r >> 16);
}

// K-pair base for VGPR slot i within a 16x32 bf16 A/B fragment (ISA 7.12.2).
__device__ __forceinline__ int kpair(int i, int half) {
    return ((i < 4) ? (2 * i) : (16 + 2 * (i - 4))) + half * 8;
}

#define FDIM 128
#define NN   256
#define DN   12
#define BB   64
#define PADI 255

#if defined(__gfx1250__) && __has_builtin(__builtin_amdgcn_tensor_load_to_lds) && \
    __has_builtin(__builtin_amdgcn_s_wait_tensorcnt)
#define USE_TDM 1
#else
#define USE_TDM 0
#endif

// ---------------------------------------------------------------------------
// GEMM: C[M,Nn] = act(A[M,K](bf16,row-major,lda) @ W(bf16,[Nn,ldb] n-major) + bias)
// grid = (M/64, Nn/128), block = 128 (4 waves).
// Wave w computes the 16x128 strip for mtile = blockIdx.x*4 + w.
// The 128-col weight strip ([128, K] bf16, padded rows) is staged in LDS once
// per block via the Tensor Data Mover (pad_interval inserts 4 dwords per row
// for power-of-two K to avoid bank conflicts).
// act: 0=none 1=leaky(0.01) 2=relu
// ---------------------------------------------------------------------------
__global__ void gemm_bf16_lds(const unsigned short* __restrict__ A, int lda,
                              const unsigned short* __restrict__ Wt, int ldb,
                              const float* __restrict__ bias,
                              float* __restrict__ Cf, unsigned short* __restrict__ Cb,
                              int M, int Nn, int K, int act,
                              int ldsStride /*elems*/, int padCode /*-1 = no pad*/) {
    extern __shared__ unsigned short ldsW[];
    const int tid = threadIdx.x;
    const int wave = tid >> 5, lane = tid & 31;
    const int half = lane >> 4, i16 = lane & 15;
    const int gy = blockIdx.y;

    // ---- stage weight strip [128 x K] into LDS ----
#if USE_TDM
    if (wave == 0) {
        unsigned long long ga =
            (unsigned long long)(size_t)(Wt + (size_t)gy * 128 * ldb);
        unsigned ldsAddr = (unsigned)(size_t)(&ldsW[0]); // low 32b = LDS offset
        tdm_v4u g0;
        g0[0] = 1u;                                   // count=1, user descriptor
        g0[1] = ldsAddr;                              // lds_addr
        g0[2] = (unsigned)ga;                         // global_addr[31:0]
        g0[3] = (unsigned)(ga >> 32) | (2u << 30);    // global_addr[56:32] | type=2
        unsigned w0 = (1u << 16);                     // data_size = 2 bytes
        if (padCode >= 0)
            w0 |= (1u << 20) | ((unsigned)padCode << 22) | (3u << 25); // pad 4 dwords
        tdm_v8i g1;
        g1[0] = (int)w0;
        g1[1] = (int)(((unsigned)K & 0xFFFFu) << 16);            // tensor_dim0[15:0]
        g1[2] = (int)((((unsigned)K >> 16) & 0xFFFFu) | (128u << 16)); // td0 hi | td1 lo
        g1[3] = (int)(((unsigned)K & 0xFFFFu) << 16);            // tile_dim0 = K
        g1[4] = (int)128;                                        // tile_dim1 = 128
        g1[5] = (int)(unsigned)ldb;                              // tensor_dim0_stride
        g1[6] = 0;
        g1[7] = 0;
        tdm_v4i gz4 = {0, 0, 0, 0};
        tdm_v8i gz8 = {0, 0, 0, 0, 0, 0, 0, 0};
        __builtin_amdgcn_tensor_load_to_lds(g0, g1, gz4, gz4, gz8, 0);
        __builtin_amdgcn_s_wait_tensorcnt((unsigned short)0);
    }
#else
    {
        const unsigned int* src =
            (const unsigned int*)(Wt + ((size_t)gy * 128 + tid) * ldb);
        unsigned int* dst = (unsigned int*)(ldsW + (size_t)tid * ldsStride);
        for (int k = 0; k < (K >> 1); ++k) dst[k] = src[k];
    }
#endif
    __syncthreads();

    const int mtile = blockIdx.x * 4 + wave;
    const size_t arow = (size_t)(mtile * 16 + i16) * lda;
    const unsigned int* ldsU = (const unsigned int*)ldsW;
    const int ldsD = ldsStride >> 1; // dwords per LDS row (multiple of 4)

    v8f acc[8];
#pragma unroll
    for (int nt = 0; nt < 8; ++nt) acc[nt] = (v8f){};

    for (int k0 = 0; k0 < K; k0 += 32) {
        Frag a;
        // 16B-aligned: lda, k0, 8*half are all multiples of 8 elems
        a.q[0] = *(const v4u*)(A + arow + k0 + 8 * half);
        a.q[1] = *(const v4u*)(A + arow + k0 + 16 + 8 * half);
#pragma unroll
        for (int nt = 0; nt < 8; ++nt) {
            const int col = nt * 16 + i16;
            const int baseD = col * ldsD + (k0 >> 1);
            Frag b;
            b.q[0] = *(const v4u*)&ldsU[baseD + 4 * half];       // ds_load_b128
            b.q[1] = *(const v4u*)&ldsU[baseD + 8 + 4 * half];   // ds_load_b128
            acc[nt] = __builtin_amdgcn_wmma_f32_16x16x32_bf16(
                false, a.v, false, b.v, (short)0, acc[nt], false, false);
        }
    }

#pragma unroll
    for (int nt = 0; nt < 8; ++nt) {
#pragma unroll
        for (int r = 0; r < 8; ++r) {
            int m = mtile * 16 + r + half * 8;
            int n = gy * 128 + nt * 16 + i16;
            if (m < M) {
                float v = acc[nt][r] + (bias ? bias[n] : 0.f);
                if (act == 1) v = (v >= 0.f) ? v : 0.01f * v;
                else if (act == 2) v = fmaxf(v, 0.f);
                if (Cf) Cf[(size_t)m * Nn + n] = v;
                if (Cb) Cb[(size_t)m * Nn + n] = f2bf(v);
            }
        }
    }
}

// ---------------------------------------------------------------------------
// Per-site self-attention over D=12 neighbors, head dim 128.
// One wave per (b,n) site. Writes softmax(QK^T/s)V as bf16 into
// concatOut[:, colOff:colOff+128] (ld = 384).
// ---------------------------------------------------------------------------
__global__ void attn_site(const unsigned short* __restrict__ Qb,
                          const unsigned short* __restrict__ Kb,
                          const unsigned short* __restrict__ Vb,
                          unsigned short* __restrict__ concatOut,
                          int colOff, float scale) {
    const size_t base = (size_t)blockIdx.x * DN;
    const int lane = threadIdx.x;
    const int half = lane >> 4, i16 = lane & 15;

    __shared__ float sS[16][16];
    __shared__ unsigned short wb[16][32]; // probs as bf16, K padded to 32

    // s = Q K^T  (contraction over 128 features)
    v8f acc = {};
    for (int k0 = 0; k0 < FDIM; k0 += 32) {
        Frag a, b;
        a.q[0] = *(const v4u*)(Qb + (base + i16) * FDIM + k0 + 8 * half);
        a.q[1] = *(const v4u*)(Qb + (base + i16) * FDIM + k0 + 16 + 8 * half);
        b.q[0] = *(const v4u*)(Kb + (base + i16) * FDIM + k0 + 8 * half);
        b.q[1] = *(const v4u*)(Kb + (base + i16) * FDIM + k0 + 16 + 8 * half);
        acc = __builtin_amdgcn_wmma_f32_16x16x32_bf16(false, a.v, false, b.v,
                                                      (short)0, acc, false, false);
    }
#pragma unroll
    for (int r = 0; r < 8; ++r) sS[r + half * 8][i16] = acc[r] * scale;

#pragma unroll
    for (int i = 0; i < 8; ++i) ((unsigned int*)&wb[0][0])[lane + 32 * i] = 0u;
    __syncthreads();

    if (lane < DN) { // row-wise softmax over the 12 valid keys
        float mx = -1e30f;
        for (int m = 0; m < DN; ++m) mx = fmaxf(mx, sS[lane][m]);
        float e[DN]; float sum = 0.f;
        for (int m = 0; m < DN; ++m) { e[m] = __expf(sS[lane][m] - mx); sum += e[m]; }
        float inv = 1.f / sum;
        for (int m = 0; m < DN; ++m) wb[lane][m] = f2bf(e[m] * inv);
    }
    __syncthreads();

    // out = P @ V : contraction over 12 keys (padded to 32 with zeros in A)
    Frag a;
    a.q[0] = *(const v4u*)&wb[i16][8 * half];        // ds_load_b128
    a.q[1] = *(const v4u*)&wb[i16][16 + 8 * half];   // ds_load_b128

    for (int nt = 0; nt < 8; ++nt) {
        Frag b;
#pragma unroll
        for (int i = 0; i < 8; ++i) {
            int kb = kpair(i, half);
            unsigned int lo = Vb[(base + kb) * FDIM + nt * 16 + i16];
            unsigned int hi = Vb[(base + kb + 1) * FDIM + nt * 16 + i16];
            b.u[i] = lo | (hi << 16);
        }
        v8f c = {};
        c = __builtin_amdgcn_wmma_f32_16x16x32_bf16(false, a.v, false, b.v,
                                                    (short)0, c, false, false);
#pragma unroll
        for (int r = 0; r < 8; ++r) {
            int m = r + half * 8;
            if (m < DN)
                concatOut[(base + m) * 384 + colOff + nt * 16 + i16] = f2bf(c[r]);
        }
    }
}

// ---------------------------------------------------------------------------
// Helpers: packing / gathers / elementwise
// ---------------------------------------------------------------------------
__global__ void pack_weight(const float* __restrict__ W, unsigned short* __restrict__ Wt,
                            int K, int Nn) {
    int n = blockIdx.x, k = threadIdx.x;
    float v = (k < K) ? W[(size_t)k * Nn + n] : 0.f;
    Wt[(size_t)n * blockDim.x + k] = f2bf(v);
}

__global__ void pack_rows(const float* __restrict__ X, unsigned short* __restrict__ Xb,
                          int Fin) {
    size_t r = blockIdx.x; int k = threadIdx.x;
    float v = (k < Fin) ? X[r * Fin + k] : 0.f;
    Xb[r * blockDim.x + k] = f2bf(v);
}

__global__ void gather_feats(const float* __restrict__ feats, const int* __restrict__ deg,
                             unsigned short* __restrict__ out, int Fin) {
    size_t r = blockIdx.x; int k = threadIdx.x;
    int b = (int)(r / (NN * DN));
    int idx = deg[r];
    float v = (k < Fin) ? feats[((size_t)b * NN + idx) * Fin + k] : 0.f;
    out[r * blockDim.x + k] = f2bf(v);
}

__global__ void gather_rows128(const float* __restrict__ actF, const int* __restrict__ deg,
                               float* __restrict__ nbF, unsigned short* __restrict__ nbB) {
    size_t r = blockIdx.x; int f = threadIdx.x;
    int b = (int)(r / (NN * DN));
    int idx = deg[r];
    float v = actF[((size_t)b * NN + idx) * FDIM + f];
    nbF[r * FDIM + f] = v;
    nbB[r * FDIM + f] = f2bf(v);
}

__device__ __forceinline__ float blockReduceSum(float v, float* red) {
    int t = threadIdx.x;
    red[t] = v; __syncthreads();
    for (int s = 64; s > 0; s >>= 1) { if (t < s) red[t] += red[t + s]; __syncthreads(); }
    float r = red[0]; __syncthreads();
    return r;
}

__global__ void score_ctx(const float* __restrict__ atomF, const float* __restrict__ nbF,
                          const float* __restrict__ attF, const int* __restrict__ atomDeg,
                          const float* __restrict__ alignW, const float* __restrict__ alignB,
                          float* __restrict__ ctxF, unsigned short* __restrict__ ctxB) {
    __shared__ float red[128];
    __shared__ float wj[DN];
    size_t s = blockIdx.x; int f = threadIdx.x;
    float afv = atomF[s * FDIM + f];
    float p1 = blockReduceSum(alignW[f] * afv, red);
    float w2 = alignW[FDIM + f];
    float sc[DN];
    for (int j = 0; j < DN; ++j)
        sc[j] = p1 + blockReduceSum(w2 * nbF[(s * DN + j) * FDIM + f], red) + alignB[0];
    if (f == 0) {
        float mx = -1e30f; float am[DN];
        for (int j = 0; j < DN; ++j) {
            int dg = atomDeg[s * DN + j];
            float v = sc[j]; v = (v >= 0.f) ? v : 0.01f * v;   // leaky
            if (dg == PADI) v += -9e8f;                         // softmax mask
            am[j] = (dg != PADI) ? 1.f : 0.f;                   // attend mask
            sc[j] = v; mx = fmaxf(mx, v);
        }
        float sum = 0.f;
        for (int j = 0; j < DN; ++j) { sc[j] = __expf(sc[j] - mx); sum += sc[j]; }
        for (int j = 0; j < DN; ++j) wj[j] = sc[j] / sum * am[j];
    }
    __syncthreads();
    float c = 0.f;
    for (int j = 0; j < DN; ++j) c += wj[j] * attF[(s * DN + j) * FDIM + f];
    c = (c > 0.f) ? c : (__expf(c) - 1.f);                      // elu
    ctxF[s * FDIM + f] = c;
    ctxB[s * FDIM + f] = f2bf(c);
}

__global__ void gru_gates(const float* __restrict__ gi, const float* __restrict__ gh,
                          float* __restrict__ atomF, unsigned short* __restrict__ atomB,
                          float* __restrict__ actF, unsigned short* __restrict__ actB) {
    size_t s = blockIdx.x; int f = threadIdx.x;
    float ir = gi[s * 384 + f], iz = gi[s * 384 + 128 + f], in_ = gi[s * 384 + 256 + f];
    float hr = gh[s * 384 + f], hz = gh[s * 384 + 128 + f], hn  = gh[s * 384 + 256 + f];
    float r = 1.f / (1.f + __expf(-(ir + hr)));
    float z = 1.f / (1.f + __expf(-(iz + hz)));
    float n = tanhf(in_ + r * hn);
    float hp = atomF[s * FDIM + f];
    float h = (1.f - z) * n + z * hp;
    atomF[s * FDIM + f] = h;  atomB[s * FDIM + f] = f2bf(h);
    float a = fmaxf(h, 0.f);
    actF[s * FDIM + f] = a;   actB[s * FDIM + f] = f2bf(a);
}

__global__ void colsum(const float* __restrict__ actF, float* __restrict__ molF,
                       float* __restrict__ actMol) {
    int b = blockIdx.x, f = threadIdx.x;
    float s = 0.f;
    for (int n = 0; n < NN; ++n) s += actF[((size_t)b * NN + n) * FDIM + f];
    molF[b * FDIM + f] = s;
    actMol[b * FDIM + f] = fmaxf(s, 0.f);
}

__global__ void mol_score_ctx(const float* __restrict__ actMol, const float* __restrict__ actF,
                              const float* __restrict__ molAtt,
                              const float* __restrict__ alignW, const float* __restrict__ alignB,
                              float* __restrict__ molCtx) {
    __shared__ float wn[NN];
    __shared__ float tmp[NN];
    int b = blockIdx.x, t = threadIdx.x;
    const float* am = actMol + b * FDIM;
    const float* av = actF + ((size_t)b * NN + t) * FDIM;
    float sc = alignB[0];
    for (int k = 0; k < FDIM; ++k) sc += alignW[k] * am[k] + alignW[FDIM + k] * av[k];
    sc = (sc >= 0.f) ? sc : 0.01f * sc;
    tmp[t] = sc; __syncthreads();
    for (int s2 = 128; s2 > 0; s2 >>= 1) { if (t < s2) tmp[t] = fmaxf(tmp[t], tmp[t + s2]); __syncthreads(); }
    float mx = tmp[0]; __syncthreads();
    float e = __expf(sc - mx);
    tmp[t] = e; __syncthreads();
    for (int s2 = 128; s2 > 0; s2 >>= 1) { if (t < s2) tmp[t] += tmp[t + s2]; __syncthreads(); }
    wn[t] = e / tmp[0]; __syncthreads();
    if (t < FDIM) {
        float c = 0.f;
        for (int n = 0; n < NN; ++n) c += wn[n] * molAtt[((size_t)b * NN + n) * FDIM + t];
        c = (c > 0.f) ? c : (__expf(c) - 1.f);
        molCtx[b * FDIM + t] = c;
    }
}

__global__ void mol_gru(const float* __restrict__ ctx, float* __restrict__ molF,
                        float* __restrict__ actMol,
                        const float* __restrict__ Wih, const float* __restrict__ Whh,
                        const float* __restrict__ bih, const float* __restrict__ bhh) {
    int b = blockIdx.x, f = threadIdx.x;
    float gi[3], gh[3];
    for (int g = 0; g < 3; ++g) { gi[g] = bih[g * 128 + f]; gh[g] = bhh[g * 128 + f]; }
    const float* x = ctx + b * FDIM;
    const float* h = molF + b * FDIM;
    for (int k = 0; k < FDIM; ++k) {
        float xv = x[k], hv = h[k];
        for (int g = 0; g < 3; ++g) {
            gi[g] += xv * Wih[k * 384 + g * 128 + f];
            gh[g] += hv * Whh[k * 384 + g * 128 + f];
        }
    }
    float r = 1.f / (1.f + __expf(-(gi[0] + gh[0])));
    float z = 1.f / (1.f + __expf(-(gi[1] + gh[1])));
    float n = tanhf(gi[2] + r * gh[2]);
    float hp = h[f];
    __syncthreads();
    float hn = (1.f - z) * n + z * hp;
    molF[b * FDIM + f] = hn;
    actMol[b * FDIM + f] = fmaxf(hn, 0.f);
}

__global__ void out_kernel(const float* __restrict__ molF, const float* __restrict__ outW,
                           const float* __restrict__ outB, float* __restrict__ out) {
    int b = threadIdx.x;
    float s = outB[0];
    for (int k = 0; k < FDIM; ++k) s += molF[b * FDIM + k] * outW[k];
    out[b] = s;
}

// ---------------------------------------------------------------------------
// Host orchestration
// ---------------------------------------------------------------------------
extern "C" void kernel_launch(void* const* d_in, const int* in_sizes, int n_in,
                              void* d_out, int out_size, void* d_ws, size_t ws_size,
                              hipStream_t stream) {
    (void)in_sizes; (void)n_in; (void)out_size; (void)ws_size;
    const size_t S = (size_t)BB * NN;     // 16384 sites
    const size_t R1 = S * DN;             // 196608 neighbor rows
    const size_t SL = 64;                 // slack rows for fragment over-read

    const float* feats[3] = { (const float*)d_in[0], (const float*)d_in[1], (const float*)d_in[2] };
    const int*   degs[3]  = { (const int*)d_in[3], (const int*)d_in[4], (const int*)d_in[5] };
    const int FinT[3]  = { 39, 10, 6 };
    const int KpadT[3] = { 64, 32, 32 };

    char* base = (char*)d_ws; size_t off = 0;
    auto alloc = [&](size_t bytes) -> void* {
        off = (off + 255) & ~(size_t)255;
        void* p = base + off; off += bytes; return p;
    };

    unsigned short* wtQKV[3][3];
    for (int t = 0; t < 3; ++t)
        for (int p = 0; p < 3; ++p)
            wtQKV[t][p] = (unsigned short*)alloc((size_t)128 * KpadT[t] * 2);
    unsigned short* wtFc     = (unsigned short*)alloc(128 * 64 * 2);
    unsigned short* wtConcat = (unsigned short*)alloc(128 * 384 * 2);
    unsigned short* wtAttend[3]; unsigned short* wtWih[3]; unsigned short* wtWhh[3];
    for (int d = 0; d < 3; ++d) {
        wtAttend[d] = (unsigned short*)alloc(128 * 128 * 2);
        wtWih[d]    = (unsigned short*)alloc(384 * 128 * 2);
        wtWhh[d]    = (unsigned short*)alloc(384 * 128 * 2);
    }
    unsigned short* wtMolAtt = (unsigned short*)alloc(128 * 128 * 2);

    unsigned short* afB = (unsigned short*)alloc(S * 64 * 2);
    unsigned short* gath[3];
    for (int t = 0; t < 3; ++t) gath[t] = (unsigned short*)alloc((R1 + SL) * KpadT[t] * 2);
    unsigned short* Qb = (unsigned short*)alloc((R1 + SL) * FDIM * 2);
    unsigned short* Kb = (unsigned short*)alloc((R1 + SL) * FDIM * 2);
    unsigned short* Vb = (unsigned short*)alloc((R1 + SL) * FDIM * 2);
    unsigned short* concatIn = (unsigned short*)alloc((R1 + SL) * 384 * 2);
    float*          nbF   = (float*)alloc(R1 * FDIM * 4);
    unsigned short* nbB   = (unsigned short*)alloc((R1 + SL) * FDIM * 2);
    float*          atomF = (float*)alloc(S * FDIM * 4);
    unsigned short* atomB = (unsigned short*)alloc(S * FDIM * 2);
    float*          attF  = (float*)alloc(R1 * FDIM * 4);
    float*          ctxF  = (float*)alloc(S * FDIM * 4);
    unsigned short* ctxB  = (unsigned short*)alloc(S * FDIM * 2);
    float*          giF   = (float*)alloc(S * 384 * 4);
    float*          ghF   = (float*)alloc(S * 384 * 4);
    float*          actF  = (float*)alloc(S * FDIM * 4);
    unsigned short* actB  = (unsigned short*)alloc(S * FDIM * 2);
    float* molF   = (float*)alloc(BB * FDIM * 4);
    float* actMol = (float*)alloc(BB * FDIM * 4);
    float* molAtt = (float*)alloc(S * FDIM * 4);
    float* molCtx = (float*)alloc(BB * FDIM * 4);

    auto gemm = [&](const unsigned short* A, int lda, const unsigned short* Wt, int ldb,
                    const float* bias, float* Cf, unsigned short* Cb,
                    int M, int Nn, int K, int act) {
        int padCode = (K == 32) ? 3 : (K == 64) ? 4 : (K == 128) ? 5 : -1;
        int ldsStride = K + ((padCode >= 0) ? 8 : 0); // elems; +16B/row kills conflicts
        size_t shmem = (size_t)128 * ldsStride * 2;
        dim3 g((unsigned)(M / 64), (unsigned)(Nn / 128));
        gemm_bf16_lds<<<g, 128, shmem, stream>>>(A, lda, Wt, ldb, bias, Cf, Cb,
                                                 M, Nn, K, act, ldsStride, padCode);
    };

    // ---- pack weights to transposed bf16 ----
    for (int t = 0; t < 3; ++t)
        for (int p = 0; p < 3; ++p)
            pack_weight<<<128, KpadT[t], 0, stream>>>((const float*)d_in[6 + t * 6 + 2 * p],
                                                      wtQKV[t][p], FinT[t], 128);
    pack_weight<<<128, 64, 0, stream>>>((const float*)d_in[24], wtFc, 39, 128);
    pack_weight<<<128, 384, 0, stream>>>((const float*)d_in[26], wtConcat, 384, 128);
    for (int d = 0; d < 3; ++d) {
        pack_weight<<<128, 128, 0, stream>>>((const float*)d_in[30] + (size_t)d * 128 * 128,
                                             wtAttend[d], 128, 128);
        pack_weight<<<384, 128, 0, stream>>>((const float*)d_in[32] + (size_t)d * 128 * 384,
                                             wtWih[d], 128, 384);
        pack_weight<<<384, 128, 0, stream>>>((const float*)d_in[33] + (size_t)d * 128 * 384,
                                             wtWhh[d], 128, 384);
    }
    pack_weight<<<128, 128, 0, stream>>>((const float*)d_in[38], wtMolAtt, 128, 128);

    // ---- gathered-neighbor self-attention (atom/bond/angle) -> concat buffer ----
    for (int t = 0; t < 3; ++t) {
        gather_feats<<<(unsigned)R1, KpadT[t], 0, stream>>>(feats[t], degs[t], gath[t], FinT[t]);
        unsigned short* QKV[3] = { Qb, Kb, Vb };
        for (int p = 0; p < 3; ++p)
            gemm(gath[t], KpadT[t], wtQKV[t][p], KpadT[t],
                 (const float*)d_in[6 + t * 6 + 2 * p + 1],
                 nullptr, QKV[p], (int)R1, 128, KpadT[t], 0);
        attn_site<<<(unsigned)S, 32, 0, stream>>>(Qb, Kb, Vb, concatIn, t * 128,
                                                  1.0f / sqrtf((float)FinT[t]));
    }

    // ---- atom_feature = leaky(af @ atom_fc_W + b) ----
    pack_rows<<<(unsigned)S, 64, 0, stream>>>(feats[0], afB, 39);
    gemm(afB, 64, wtFc, 64, (const float*)d_in[25], atomF, atomB, (int)S, 128, 64, 1);

    // ---- neighbor = concat(...) @ concat_W + b ----
    gemm(concatIn, 384, wtConcat, 384, (const float*)d_in[27], nbF, nbB, (int)R1, 128, 384, 0);

    // ---- radius loop ----
    for (int d = 0; d < 3; ++d) {
        if (d > 0)
            gather_rows128<<<(unsigned)R1, 128, 0, stream>>>(actF, degs[0], nbF, nbB);
        gemm(nbB, 128, wtAttend[d], 128, (const float*)d_in[31] + (size_t)d * 128,
             attF, nullptr, (int)R1, 128, 128, 0);
        score_ctx<<<(unsigned)S, 128, 0, stream>>>(atomF, nbF, attF, degs[0],
                                                   (const float*)d_in[28] + (size_t)d * 256,
                                                   (const float*)d_in[29] + d, ctxF, ctxB);
        gemm(ctxB, 128, wtWih[d], 128, (const float*)d_in[34] + (size_t)d * 384,
             giF, nullptr, (int)S, 384, 128, 0);
        gemm(atomB, 128, wtWhh[d], 128, (const float*)d_in[35] + (size_t)d * 384,
             ghF, nullptr, (int)S, 384, 128, 0);
        gru_gates<<<(unsigned)S, 128, 0, stream>>>(giF, ghF, atomF, atomB, actF, actB);
    }

    // ---- molecule-level readout ----
    colsum<<<BB, 128, 0, stream>>>(actF, molF, actMol);
    gemm(actB, 128, wtMolAtt, 128, (const float*)d_in[39], molAtt, nullptr, (int)S, 128, 128, 0);
    for (int t = 0; t < 2; ++t) {
        mol_score_ctx<<<BB, 256, 0, stream>>>(actMol, actF, molAtt,
                                              (const float*)d_in[36], (const float*)d_in[37],
                                              molCtx);
        mol_gru<<<BB, 128, 0, stream>>>(molCtx, molF, actMol,
                                        (const float*)d_in[40], (const float*)d_in[41],
                                        (const float*)d_in[42], (const float*)d_in[43]);
    }
    out_kernel<<<1, BB, 0, stream>>>(molF, (const float*)d_in[44], (const float*)d_in[45],
                                     (float*)d_out);
}
